// HERALD_34660386079385
// MI455X (gfx1250) — compile-verified, matrix operand order checked by reading, not run on hardware
//
#include <hip/hip_runtime.h>

// ---------------------------------------------------------------------------
// Problem dimensions (fixed by the reference)
// ---------------------------------------------------------------------------
#define VN 4096
#define EN 2048
#define FN 512
#define HN 256
#define PI_F 3.14159265358979f

// ---------------------------------------------------------------------------
// Types / bf16 helpers (bf16 kept as raw u16 bits; reinterpret at WMMA boundary)
// ---------------------------------------------------------------------------
typedef __attribute__((ext_vector_type(16))) __bf16          v16bf;
typedef __attribute__((ext_vector_type(16))) unsigned short  u16x16;
typedef __attribute__((ext_vector_type(8)))  unsigned short  u16x8;
typedef __attribute__((ext_vector_type(8)))  float           v8f;
typedef __attribute__((ext_vector_type(4)))  unsigned int    uint32x4;
typedef __attribute__((ext_vector_type(4)))  int             int32x4;
typedef __attribute__((ext_vector_type(8)))  int             int32x8;

#if defined(__has_builtin)
#if __has_builtin(__builtin_amdgcn_tensor_load_to_lds) && \
    __has_builtin(__builtin_amdgcn_s_wait_tensorcnt)
#define HAVE_TDM 1
#endif
#endif

static __device__ __forceinline__ unsigned short f2bf_bits(float f) {
    unsigned u = __float_as_uint(f);
    unsigned r = u + 0x7FFFu + ((u >> 16) & 1u);   // round-to-nearest-even
    return (unsigned short)(r >> 16);
}

static __device__ __forceinline__ v16bf load_frag(const unsigned short* p) {
    u16x8 lo = *(const u16x8*)p;
    u16x8 hi = *(const u16x8*)(p + 16);
    union { u16x16 u; v16bf b; } cv;
    cv.u = __builtin_shufflevector(lo, hi, 0,1,2,3,4,5,6,7,8,9,10,11,12,13,14,15);
    return cv.b;
}

// ---------------------------------------------------------------------------
// GEMM tile geometry
// ---------------------------------------------------------------------------
#define BM 128
#define BN 128
#define BK 32
#define LDSK 40   // padded LDS row stride in halves (80B = 16B-aligned, conflict-free)

enum { EPI_STORE = 0, EPI_ROWDIV = 1, EPI_SCALE16 = 2, EPI_EXPDIST = 3, EPI_GBLEND = 4 };

#ifdef HAVE_TDM
// Issue one TDM transfer: 128 rows x 32 bf16 tile, row stride `ld` halves,
// LDS padding 4 DWORDs every 16 DWORDs -> 80B LDS row stride (== LDSK).
static __device__ __forceinline__ void tdm_load_tile(const unsigned short* g,
                                                     unsigned ldsaddr, int ld) {
    unsigned long long ga = (unsigned long long)(size_t)g;
    uint32x4 g0;
    g0[0] = 1u;                                            // count=1, user mode
    g0[1] = ldsaddr;                                       // LDS byte address
    g0[2] = (unsigned)ga;                                  // global addr lo
    g0[3] = (unsigned)((ga >> 32) & 0x01FFFFFFull) | (2u << 30);  // hi + type=2
    int32x8 g1;
    g1[0] = (1 << 16) | (1 << 20) | (3 << 22) | (3 << 25); // 2B data, pad 4dw/16dw
    g1[1] = (ld & 0xFFFF) << 16;                           // tensor_dim0 lo
    g1[2] = ((unsigned)ld >> 16);                          // tensor_dim0 hi (dim1 lo=0)
    g1[3] = (int)((1u << 20) >> 16) | (32 << 16);          // tensor_dim1=1M, tile_dim0=32
    g1[4] = 128;                                           // tile_dim1=128, tile_dim2=0
    g1[5] = ld;                                            // tensor_dim0_stride lo
    g1[6] = 0;
    g1[7] = 0;
    int32x4 g2; g2[0] = 1; g2[1] = 1; g2[2] = 0; g2[3] = 0;       // dim2=dim3=1
    int32x4 g3; g3[0] = 0; g3[1] = (1 << 16); g3[2] = 0; g3[3] = 0; // dim4=1
#if __clang_major__ >= 23
    int32x8 g4 = {0, 0, 0, 0, 0, 0, 0, 0};
    __builtin_amdgcn_tensor_load_to_lds(g0, g1, g2, g3, g4, 0);
#else
    __builtin_amdgcn_tensor_load_to_lds(g0, g1, g2, g3, 0);
#endif
}
#endif

// ---------------------------------------------------------------------------
// Tiled WMMA GEMM:  C(f32, MxN) = epilogue( A(bf16, MxK row-major) * B^T )
//   B is N x K row-major (all operands pre-arranged this way).
// 256 threads = 8 wave32 waves; block tile 128x128x32; wave tile 64x32.
// Double-buffered LDS; staging via TDM (async tensor DMA) when available.
// ---------------------------------------------------------------------------
template <int EPI>
__global__ __launch_bounds__(256)
void k_gemm(const unsigned short* __restrict__ A, int lda,
            const unsigned short* __restrict__ B, int ldb,
            float* __restrict__ C, int ldc,
            int M, int N, int K,
            const float* __restrict__ rowAux,   // deg / a / invDV
            const float* __restrict__ colAux,   // c / invDV
            const float* __restrict__ sclF,     // w_o_b
            const int*   __restrict__ sclI,     // sigma or num
            const float* __restrict__ Gin)      // old G for blend
{
    __shared__ unsigned short As[2][BM * LDSK];
    __shared__ unsigned short Bs[2][BN * LDSK];

    const int tid  = threadIdx.x;
    const int wid  = tid >> 5;
    const int lane = tid & 31;
    const int bM   = blockIdx.y * BM;
    const int bN   = blockIdx.x * BN;
    const int wm   = (wid & 1) * 64;
    const int wn   = (wid >> 1) * 32;
    const int T    = K / BK;

    v8f acc[4][2];
#pragma unroll
    for (int i = 0; i < 4; ++i)
#pragma unroll
        for (int j = 0; j < 2; ++j)
#pragma unroll
            for (int r = 0; r < 8; ++r) acc[i][j][r] = 0.0f;

    const int mrow  = lane & 15;
    const int khalf = (lane >> 4) * 8;

    const int ar = tid >> 1;            // 0..127 : tile row handled by this thread
    const int ak = (tid & 1) * 16;      // 0 | 16 halves
    const unsigned short* gA = A + (size_t)(bM + ar) * lda + ak;
    const unsigned short* gB = B + (size_t)(bN + ar) * ldb + ak;

#ifdef HAVE_TDM
    if (wid == 0) {
        tdm_load_tile(A + (size_t)bM * lda, (unsigned)(size_t)&As[0][0], lda);
        tdm_load_tile(B + (size_t)bN * ldb, (unsigned)(size_t)&Bs[0][0], ldb);
    }
#else
    {   // prologue: stage tile 0 into buffer 0
        uint4 a0 = *(const uint4*)gA;       uint4 a1 = *(const uint4*)(gA + 8);
        uint4 b0 = *(const uint4*)gB;       uint4 b1 = *(const uint4*)(gB + 8);
        *(uint4*)&As[0][ar * LDSK + ak]     = a0;
        *(uint4*)&As[0][ar * LDSK + ak + 8] = a1;
        *(uint4*)&Bs[0][ar * LDSK + ak]     = b0;
        *(uint4*)&Bs[0][ar * LDSK + ak + 8] = b1;
    }
#endif

    for (int t = 0; t < T; ++t) {
        const int cur = t & 1;
        const int nxt = cur ^ 1;
#ifdef HAVE_TDM
        __syncthreads();                 // buffer `nxt` fully consumed
        if (wid == 0) {
            if (t + 1 < T) {
                tdm_load_tile(A + (size_t)bM * lda + (size_t)(t + 1) * BK,
                              (unsigned)(size_t)&As[nxt][0], lda);
                tdm_load_tile(B + (size_t)bN * ldb + (size_t)(t + 1) * BK,
                              (unsigned)(size_t)&Bs[nxt][0], ldb);
                __builtin_amdgcn_s_wait_tensorcnt(2);   // tile t landed
            } else {
                __builtin_amdgcn_s_wait_tensorcnt(0);
            }
        }
        __syncthreads();                 // tile t visible to all waves
#else
        __syncthreads();                 // tile t stores visible; `nxt` free
        uint4 a0 = {}, a1 = {}, b0 = {}, b1 = {};
        const bool more = (t + 1 < T);
        if (more) {
            const unsigned short* pA = gA + (size_t)(t + 1) * BK;
            const unsigned short* pB = gB + (size_t)(t + 1) * BK;
            a0 = *(const uint4*)pA;  a1 = *(const uint4*)(pA + 8);
            b0 = *(const uint4*)pB;  b1 = *(const uint4*)(pB + 8);
            if (t + 2 < T) {
                __builtin_prefetch(pA + BK, 0, 1);
                __builtin_prefetch(pB + BK, 0, 1);
            }
        }
#endif
        // ---- compute on buffer `cur`
        v16bf afrag[4], bfrag[2];
#pragma unroll
        for (int j = 0; j < 2; ++j)
            bfrag[j] = load_frag(&Bs[cur][(wn + j * 16 + mrow) * LDSK + khalf]);
#pragma unroll
        for (int i = 0; i < 4; ++i) {
            afrag[i] = load_frag(&As[cur][(wm + i * 16 + mrow) * LDSK + khalf]);
#pragma unroll
            for (int j = 0; j < 2; ++j)
                acc[i][j] = __builtin_amdgcn_wmma_f32_16x16x32_bf16(
                    false, afrag[i], false, bfrag[j], (short)0, acc[i][j],
                    false, false);
        }
#ifndef HAVE_TDM
        if (more) {
            *(uint4*)&As[nxt][ar * LDSK + ak]     = a0;
            *(uint4*)&As[nxt][ar * LDSK + ak + 8] = a1;
            *(uint4*)&Bs[nxt][ar * LDSK + ak]     = b0;
            *(uint4*)&Bs[nxt][ar * LDSK + ak + 8] = b1;
        }
#endif
    }

    // ---- epilogue: C/D layout -> VGPR r : row = r + (lane<16 ? 0 : 8), col = lane&15
    const int radd = (lane >> 4) * 8;
    const int cidx = lane & 15;

    float inv2s = 0.0f, bias = 0.0f, theta = 0.0f;
    if (EPI == EPI_EXPDIST) {
        float sg = (float)sclI[0];
        inv2s = 1.0f / (2.0f * sg * sg);
        bias  = sclF[0];
    }
    if (EPI == EPI_GBLEND) {
        float nm = (float)sclI[0];
        theta = 1.0f - (1.0f - 0.01f) * (cosf(PI_F * (nm - 1.0f) * 0.1f) + 1.0f) * 0.5f;
    }

#pragma unroll
    for (int i = 0; i < 4; ++i) {
#pragma unroll
        for (int j = 0; j < 2; ++j) {
            const int col = bN + wn + j * 16 + cidx;
#pragma unroll
            for (int r = 0; r < 8; ++r) {
                const int row = bM + wm + i * 16 + r + radd;
                const size_t idx = (size_t)row * ldc + col;
                float v = acc[i][j][r];
                if (EPI == EPI_STORE) {
                    C[idx] = v;
                } else if (EPI == EPI_ROWDIV) {
                    C[idx] = v / rowAux[row];
                } else if (EPI == EPI_SCALE16) {
                    C[idx] = v * 0.0625f;               // 1/sqrt(HN)
                } else if (EPI == EPI_EXPDIST) {
                    float dist = rowAux[row] + colAux[col] - 2.0f * v + bias;
                    C[idx] = expf(-dist * inv2s);
                } else {                                  // EPI_GBLEND
                    C[idx] = (1.0f - theta) * Gin[idx] +
                             theta * rowAux[row] * colAux[col] * v;
                }
            }
        }
    }
}

// ---------------------------------------------------------------------------
// Elementwise / pack / reduction kernels
// ---------------------------------------------------------------------------
__global__ __launch_bounds__(256)
void k_cvt(const float* __restrict__ in, unsigned short* __restrict__ out, int n) {
    int i = blockIdx.x * 256 + threadIdx.x;
    if (i < n) out[i] = f2bf_bits(in[i]);
}

// WbT (512 x 512, N x K): WbT[n,k] = (n<256 ? W_lin[k,n] : W_v[k,n-256])
__global__ __launch_bounds__(256)
void k_pack_WT(const float* __restrict__ wl, const float* __restrict__ wv,
               unsigned short* __restrict__ WbT) {
    int i = blockIdx.x * 256 + threadIdx.x;        // over 512*512
    if (i < FN * 2 * HN) {
        int n = i >> 9, k = i & 511;
        float x = (n < HN) ? wl[k * HN + n] : wv[k * HN + (n - HN)];
        WbT[i] = f2bf_bits(x);
    }
}

// fvb (V x H row-major) from PV columns [256..512)
__global__ __launch_bounds__(256)
void k_pack_fvb(const float* __restrict__ PV, unsigned short* __restrict__ fvb, int n) {
    int i = blockIdx.x * 256 + threadIdx.x;
    if (i < n) {
        int v = i >> 8, h = i & 255;
        fvb[i] = f2bf_bits(PV[v * (2 * HN) + HN + h]);
    }
}

// out[c] = reduce_r in[r*C + c]; mode 0: raw sum, 1: DV^-1/2 (0 if 0)
__global__ __launch_bounds__(256)
void k_colsum(const float* __restrict__ in, float* __restrict__ out,
              int R, int C, int mode) {
    int c = blockIdx.x * 256 + threadIdx.x;
    if (c >= C) return;
    float s = 0.0f;
    for (int r = 0; r < R; ++r) s += in[(size_t)r * C + c];
    out[c] = (mode == 1) ? ((s == 0.0f) ? 0.0f : rsqrtf(s)) : s;
}

// out[row] = 1/rowsum (0 if 0); one block per row
__global__ __launch_bounds__(256)
void k_rowsum_inv(const float* __restrict__ in, float* __restrict__ out, int C) {
    __shared__ float red[256];
    const int row = blockIdx.x, t = threadIdx.x;
    const float* p = in + (size_t)row * C;
    float s = 0.0f;
    for (int i = t; i < C; i += 256) s += p[i];
    red[t] = s; __syncthreads();
    for (int st = 128; st > 0; st >>= 1) {
        if (t < st) red[t] += red[t + st];
        __syncthreads();
    }
    if (t == 0) out[row] = (red[0] == 0.0f) ? 0.0f : 1.0f / red[0];
}

// row softmax of S (rows of length N), write normalized bf16
__global__ __launch_bounds__(256)
void k_softmax_bf16(const float* __restrict__ S, unsigned short* __restrict__ outB, int N) {
    __shared__ float rowb[VN];
    __shared__ float red[256];
    const int r = blockIdx.x, t = threadIdx.x;
    const float* p = S + (size_t)r * N;
    float mx = -3.4e38f;
    for (int i = t; i < N; i += 256) { float x = p[i]; rowb[i] = x; mx = fmaxf(mx, x); }
    red[t] = mx; __syncthreads();
    for (int st = 128; st > 0; st >>= 1) {
        if (t < st) red[t] = fmaxf(red[t], red[t + st]);
        __syncthreads();
    }
    mx = red[0]; __syncthreads();
    float sm = 0.0f;
    for (int i = t; i < N; i += 256) { float e = expf(rowb[i] - mx); rowb[i] = e; sm += e; }
    red[t] = sm; __syncthreads();
    for (int st = 128; st > 0; st >>= 1) {
        if (t < st) red[t] += red[t + st];
        __syncthreads();
    }
    float inv = 1.0f / red[0];
    for (int i = t; i < N; i += 256) outB[(size_t)r * N + i] = f2bf_bits(rowb[i] * inv);
}

// layernorm row (H=256, one block per row, one element per thread)
// writes bf16 (mulW ? y*w : y) and outVec[row] = sum(w * y^2)
__global__ __launch_bounds__(256)
void k_layernorm(const float* __restrict__ X, const float* __restrict__ g,
                 const float* __restrict__ b, const float* __restrict__ w,
                 unsigned short* __restrict__ outB, float* __restrict__ outVec,
                 int mulW) {
    __shared__ float red[256];
    const int row = blockIdx.x, t = threadIdx.x;
    float x = X[(size_t)row * HN + t];
    red[t] = x; __syncthreads();
    for (int st = 128; st > 0; st >>= 1) { if (t < st) red[t] += red[t + st]; __syncthreads(); }
    float mu = red[0] * (1.0f / HN); __syncthreads();
    float d = x - mu;
    red[t] = d * d; __syncthreads();
    for (int st = 128; st > 0; st >>= 1) { if (t < st) red[t] += red[t + st]; __syncthreads(); }
    float var = red[0] * (1.0f / HN); __syncthreads();
    float y  = d * rsqrtf(var + 1e-5f) * g[t] + b[t];
    float wv = w[t];
    outB[(size_t)row * HN + t] = f2bf_bits(mulW ? y * wv : y);
    red[t] = wv * y * y; __syncthreads();
    for (int st = 128; st > 0; st >>= 1) { if (t < st) red[t] += red[t + st]; __syncthreads(); }
    if (t == 0) outVec[row] = red[0];
}

// transpose + bf16 convert: in rows R x (ld ldin, columns [col0, col0+gridC*32))
// -> out[(c)*R + r] bf16
__global__ __launch_bounds__(256)
void k_transpose_cvt(const float* __restrict__ in, unsigned short* __restrict__ out,
                     int R, int ldin, int col0) {
    __shared__ float tile[32][33];
    const int c0 = blockIdx.x * 32, r0 = blockIdx.y * 32;
    const int tx = threadIdx.x, ty = threadIdx.y;
    for (int j = ty; j < 32; j += 8)
        tile[j][tx] = in[(size_t)(r0 + j) * ldin + col0 + c0 + tx];
    __syncthreads();
    for (int j = ty; j < 32; j += 8)
        out[(size_t)(c0 + j) * R + r0 + tx] = f2bf_bits(tile[tx][j]);
}

// pack H matrices for the final GEMM:
//   M1[v*E+e] = bf16(T[e*V+v] * invDE[e]) ; M2[v*E+e] = bf16(T[e*V+v])
__global__ __launch_bounds__(256)
void k_pack_M(const float* __restrict__ T, const float* __restrict__ invDE,
              unsigned short* __restrict__ M1, unsigned short* __restrict__ M2,
              int R /*E*/, int C /*V*/) {
    __shared__ float tile[32][33];
    const int c0 = blockIdx.x * 32, r0 = blockIdx.y * 32;
    const int tx = threadIdx.x, ty = threadIdx.y;
    for (int j = ty; j < 32; j += 8)
        tile[j][tx] = T[(size_t)(r0 + j) * C + c0 + tx];
    __syncthreads();
    const float ide = invDE[r0 + tx];
    for (int j = ty; j < 32; j += 8) {
        float t = tile[tx][j];
        size_t o = (size_t)(c0 + j) * R + r0 + tx;
        M1[o] = f2bf_bits(t * ide);
        M2[o] = f2bf_bits(t);
    }
}

// ---------------------------------------------------------------------------
// Launch
// ---------------------------------------------------------------------------
extern "C" void kernel_launch(void* const* d_in, const int* in_sizes, int n_in,
                              void* d_out, int out_size, void* d_ws, size_t ws_size,
                              hipStream_t stream) {
    (void)in_sizes; (void)n_in; (void)out_size; (void)ws_size;
    const float* adj   = (const float*)d_in[0];
    const float* G     = (const float*)d_in[1];
    const float* feats = (const float*)d_in[2];
    const float* W_lin = (const float*)d_in[3];
    const float* W_v   = (const float*)d_in[4];
    const float* w_o_w = (const float*)d_in[5];
    const float* w_o_b = (const float*)d_in[6];
    const float* ln1_g = (const float*)d_in[7];
    const float* ln1_b = (const float*)d_in[8];
    const float* ln2_g = (const float*)d_in[9];
    const float* ln2_b = (const float*)d_in[10];
    const int*   num   = (const int*)d_in[11];
    const int*   sigma = (const int*)d_in[12];
    float* out = (float*)d_out;

    char* base = (char*)d_ws;
    size_t off = 0;
    auto take = [&](size_t bytes) { size_t r = off; off += (bytes + 255) & ~(size_t)255; return r; };

    unsigned short* WbT   = (unsigned short*)(base + take((size_t)FN * 2 * HN * 2));
    unsigned short* PbT   = (unsigned short*)(base + take((size_t)VN * HN * 2));
    unsigned short* fvb   = (unsigned short*)(base + take((size_t)VN * HN * 2));
    unsigned short* fvbT  = (unsigned short*)(base + take((size_t)VN * HN * 2));
    float*          deg   = (float*)(base + take((size_t)EN * 4));
    float*          sMat  = (float*)(base + take((size_t)EN * HN * 4));
    unsigned short* swb   = (unsigned short*)(base + take((size_t)EN * HN * 2));
    float*          aE    = (float*)(base + take((size_t)EN * 4));
    float*          dMat  = (float*)(base + take((size_t)VN * HN * 4));
    unsigned short* db    = (unsigned short*)(base + take((size_t)VN * HN * 2));
    float*          cV    = (float*)(base + take((size_t)VN * 4));
    float*          invDE = (float*)(base + take((size_t)EN * 4));
    float*          invDV = (float*)(base + take((size_t)VN * 4));
    char* RA = base + take((size_t)VN * VN * 4);   // 64 MB region (phased reuse)
    char* RB = base + take((size_t)EN * VN * 4);   // 32 MB region (phased reuse)

    // Region A phase 1: PV f32 (8MB) | featsB (4MB) | adjT bf16 (16MB)
    float*          PV     = (float*)RA;
    unsigned short* featsB = (unsigned short*)(RA + ((size_t)VN * 2 * HN * 4));
    unsigned short* adjTb  = (unsigned short*)(RA + ((size_t)VN * 2 * HN * 4) + ((size_t)VN * FN * 2));
    // Region A phase 2: attention scores (64MB)
    float*          scores = (float*)RA;
    // Region A phase 3: M1 | M2 (16MB each)
    unsigned short* M1 = (unsigned short*)RA;
    unsigned short* M2 = (unsigned short*)(RA + (size_t)VN * EN * 2);
    // Region B: attn bf16 (32MB) -> T f32 (32MB)
    unsigned short* attnB = (unsigned short*)RB;
    float*          Tm    = (float*)RB;

    // 1. operand prep: feats->bf16, W^T pack, deg_e, adj^T (bf16)
    k_cvt<<<(VN * FN + 255) / 256, 256, 0, stream>>>(feats, featsB, VN * FN);
    k_pack_WT<<<(FN * 2 * HN + 255) / 256, 256, 0, stream>>>(W_lin, W_v, WbT);
    k_colsum<<<EN / 256, 256, 0, stream>>>(adj, deg, VN, EN, 0);
    k_transpose_cvt<<<dim3(EN / 32, VN / 32), dim3(32, 8), 0, stream>>>(adj, adjTb, VN, EN, 0);

    // 2. PV = feats @ [W_lin | W_v]   (V x 512)
    k_gemm<EPI_STORE><<<dim3((2 * HN) / BN, VN / BM), 256, 0, stream>>>(
        featsB, FN, WbT, FN, PV, 2 * HN, VN, 2 * HN, FN,
        nullptr, nullptr, nullptr, nullptr, nullptr);
    k_pack_fvb<<<(VN * HN + 255) / 256, 256, 0, stream>>>(PV, fvb, VN * HN);
    k_transpose_cvt<<<dim3(HN / 32, VN / 32), dim3(32, 8), 0, stream>>>(PV, PbT, VN, 2 * HN, 0);
    k_transpose_cvt<<<dim3(HN / 32, VN / 32), dim3(32, 8), 0, stream>>>(PV, fvbT, VN, 2 * HN, HN);

    // 3. s = (adj^T @ P) / deg     (E x H)
    k_gemm<EPI_ROWDIV><<<dim3(HN / BN, EN / BM), 256, 0, stream>>>(
        adjTb, VN, PbT, VN, sMat, HN, EN, HN, VN,
        deg, nullptr, nullptr, nullptr, nullptr);

    // 4. scores = fv @ fv^T / 16   (V x V)    [overwrites region A phase 1]
    k_gemm<EPI_SCALE16><<<dim3(VN / BN, VN / BM), 256, 0, stream>>>(
        fvb, HN, fvb, HN, scores, VN, VN, VN, HN,
        nullptr, nullptr, nullptr, nullptr, nullptr);

    // 5. attn = softmax(scores) -> bf16
    k_softmax_bf16<<<VN, 256, 0, stream>>>(scores, attnB, VN);

    // 6. d = attn @ fv   (V x H)
    k_gemm<EPI_STORE><<<dim3(HN / BN, VN / BM), 256, 0, stream>>>(
        attnB, VN, fvbT, VN, dMat, HN, VN, HN, VN,
        nullptr, nullptr, nullptr, nullptr, nullptr);

    // 7. layernorms (+ a, c vectors, bf16 packs)
    k_layernorm<<<EN, 256, 0, stream>>>(sMat, ln1_g, ln1_b, w_o_w, swb, aE, 1);
    k_layernorm<<<VN, 256, 0, stream>>>(dMat, ln2_g, ln2_b, w_o_w, db, cV, 0);

    // 8. T = exp(-(a_i + c_j - 2 (s*w)d^T + b)/(2 sigma^2))   (E x V)
    k_gemm<EPI_EXPDIST><<<dim3(VN / BN, EN / BM), 256, 0, stream>>>(
        swb, HN, db, HN, Tm, VN, EN, VN, HN,
        aE, cV, w_o_b, sigma, nullptr);

    // 9. hypergraph degree scalings
    k_rowsum_inv<<<EN, 256, 0, stream>>>(Tm, invDE, VN);
    k_colsum<<<VN / 256, 256, 0, stream>>>(Tm, invDV, EN, VN, 1);
    k_pack_M<<<dim3(VN / 32, EN / 32), dim3(32, 8), 0, stream>>>(Tm, invDE, M1, M2, EN, VN);

    // 10. G_out = (1-theta) G + theta * diag(invDV) (M1 M2^T) diag(invDV)
    k_gemm<EPI_GBLEND><<<dim3(VN / BN, VN / BM), 256, 0, stream>>>(
        M1, EN, M2, EN, out, VN, VN, VN, EN,
        invDV, invDV, nullptr, num, G);
}
